// DKBATNet_55748675502098
// MI455X (gfx1250) — matrix operand order repaired and checked by reading, not compiled, not run on hardware
//
#include <hip/hip_runtime.h>
#include <hip/hip_bf16.h>

// ---------------------------------------------------------------------------
// DKBAT 2-hop relational GAT for MI455X (gfx1250, wave32, WMMA bf16,
// async global->LDS weight staging)
// ---------------------------------------------------------------------------

#define N_NODES 100000
#define M_REL   500
#define E_EDGES 300000
#define XS_     100
#define GS_     100
#define D_      200     // HEADS*OUT

typedef __attribute__((ext_vector_type(16))) __bf16 v16bf;
typedef __attribute__((ext_vector_type(8)))  float  v8f;

__device__ __forceinline__ unsigned short f2bf(float f) {
    union { float f; unsigned u; } x; x.f = f;
    unsigned u = x.u;
    return (unsigned short)((u + 0x7FFFu + ((u >> 16) & 1u)) >> 16);
}

// Issue one per-lane 16B async copy global->LDS (tracked by ASYNCcnt).
__device__ __forceinline__ void async_copy_b128(unsigned lds_addr,
                                                const void* gaddr) {
    asm volatile("global_load_async_to_lds_b128 %0, %1, off"
                 :: "v"(lds_addr), "v"(gaddr) : "memory");
}
__device__ __forceinline__ void wait_asynccnt0() {
    asm volatile("s_wait_asynccnt 0x0" ::: "memory");
}

// -------------------- row-wise L2 normalize (x) ----------------------------
__global__ __launch_bounds__(128) void l2norm_rows(const float* __restrict__ x,
                                                   float* __restrict__ y,
                                                   int width) {
    __shared__ float s;
    int r = blockIdx.x, t = threadIdx.x;
    if (t == 0) s = 0.f;
    __syncthreads();
    float v = 0.f;
    if (t < width) { v = x[(size_t)r * width + t]; atomicAdd(&s, v * v); }
    __syncthreads();
    if (t < width) y[(size_t)r * width + t] = v / fmaxf(sqrtf(s), 1e-12f);
}

// -------------------- pack weights into WMMA B fragments -------------------
// B element (k, n) of W^T (i.e. W[n][k]):
//   lane = (n%16) + 16*(k_local>=16), half j = k_local%16
__global__ void pack_bfrag(const float* __restrict__ s0, const float* __restrict__ s1,
                           int rows0, int Dout, int K, int NT, int KC,
                           unsigned short* __restrict__ dst) {
    int total = NT * KC * 512;
    for (int i = blockIdx.x * blockDim.x + threadIdx.x; i < total;
         i += gridDim.x * blockDim.x) {
        int j    = i & 15;
        int lane = (i >> 4) & 31;
        int kc   = (i >> 9) % KC;
        int nt   = i / (KC << 9);
        int nl = lane & 15, kh = lane >> 4;
        int k = kc * 32 + kh * 16 + j;
        int n = nt * 16 + nl;
        float v = 0.f;
        if (n < Dout && k < K)
            v = (n < rows0) ? s0[(size_t)n * K + k]
                            : s1[(size_t)(n - rows0) * K + k];
        dst[i] = f2bf(v);
    }
}

// -------------------- edge GAT pass (A: logits, B: aggregate) --------------
// 2 waves / block, 32 edges / block, wave w owns M-tile w (16 edges).
// A gathered into LDS in A-fragment layout; B async-staged into LDS per
// N-tile (one copy shared by both waves), consumed via ds_load_b128.
__global__ __launch_bounds__(64) void edge_gat_pass(
    int passB,
    const int* __restrict__ rowi, const int* __restrict__ coli,
    const int* __restrict__ etype,
    const float* __restrict__ p0, int w0,
    const float* __restrict__ p1, int w1,
    const float* __restrict__ p2, int w2,
    int Kpad, int KC, const unsigned short* __restrict__ wfrag,
    int NT, int OUT, int NHEADS,
    const float* __restrict__ attin, const float* __restrict__ attout,
    float* __restrict__ ebuf, float* __restrict__ denom,
    float* __restrict__ hin, float* __restrict__ hout) {
    __shared__ __align__(32) unsigned short sA[2 * 20 * 512];   // up to K=640
    __shared__ __align__(32) unsigned short sB[20 * 512];       // one N-tile
    __shared__ float sAtt[32 * 4];

    int tid = threadIdx.x, lane = tid & 31, wv = tid >> 5;
    int ebase = blockIdx.x * 32;
    int wtot = w0 + w1 + w2;

    // ---- gather + bf16 convert + scatter into A-fragment layout in LDS ----
    for (int i = tid; i < 32 * Kpad; i += 64) {
        int el = i / Kpad, k = i - el * Kpad;
        int edge = ebase + el;
        float v = 0.f;
        if (edge < E_EDGES && k < wtot) {
            if (k < w0)            v = p0[(size_t)rowi[edge] * w0 + k];
            else if (k < w0 + w1)  v = p1[(size_t)coli[edge] * w1 + (k - w0)];
            else                   v = p2[(size_t)etype[edge] * w2 + (k - w0 - w1)];
        }
        int tile = el >> 4, m = el & 15, kc = k >> 5, kk = k & 31;
        int hi = (kk >> 4) & 1;
        int ls = m + (hi << 4);
        int j  = kk & 15;
        sA[((tile * KC + kc) << 9) + (ls << 4) + j] = f2bf(v);
    }
    for (int i = tid; i < 32 * 4; i += 64) sAtt[i] = 0.f;
    __syncthreads();

    int t = wv;
    for (int nt = 0; nt < NT; ++nt) {
        // ---- async-stage this N-tile's B fragments into LDS (KC*1024 B) ----
        {
            const unsigned short* gsrc = &wfrag[((size_t)nt * KC) << 9];
            for (int c = tid; c < KC * 64; c += 64) {   // KC*64 % 64 == 0
                unsigned lds = (unsigned)(unsigned long long)&sB[c * 8];
                async_copy_b128(lds, gsrc + c * 8);
            }
            wait_asynccnt0();
        }
        __syncthreads();

        v8f acc = {0.f, 0.f, 0.f, 0.f, 0.f, 0.f, 0.f, 0.f};
        for (int kc = 0; kc < KC; ++kc) {
            v16bf a = *(const v16bf*)&sA[((t * KC + kc) << 9) + (lane << 4)];
            v16bf b = *(const v16bf*)&sB[(kc << 9) + (lane << 4)];
            acc = __builtin_amdgcn_wmma_f32_16x16x32_bf16(
                false, a, false, b, (short)0, acc, false, false);
        }
        // C layout: n = nt*16 + lane%16 ; m = r + 8*(lane>=16)
        int n  = nt * 16 + (lane & 15);
        int mh = (lane >> 4) << 3;
        int slot = (n >= D_ ? NHEADS : 0) + ((n % D_) / OUT);
        if (!passB) {
            float attv = (n < D_) ? attin[n] : attout[n - D_];
            for (int r = 0; r < 8; ++r) {
                int el = t * 16 + r + mh;
                if (ebase + el < E_EDGES)
                    atomicAdd(&sAtt[el * 4 + slot], acc[r] * attv);
            }
        } else {
            for (int r = 0; r < 8; ++r) {
                int el = t * 16 + r + mh;
                int edge = ebase + el;
                if (edge < E_EDGES) {
                    int node = (n < D_) ? coli[edge] : rowi[edge];
                    float alpha = ebuf[(size_t)edge * 4 + slot] /
                                  denom[(size_t)node * 4 + slot];
                    float* dst = (n < D_) ? &hin[(size_t)node * D_ + n]
                                          : &hout[(size_t)node * D_ + (n - D_)];
                    unsafeAtomicAdd(dst, alpha * acc[r]);
                }
            }
        }
        __syncthreads();    // protect sB before next N-tile's staging
    }

    if (!passB) {
        int HA = 2 * NHEADS;
        for (int i = tid; i < 32 * HA; i += 64) {
            int el = i / HA, slot = i - el * HA;
            int edge = ebase + el;
            if (edge < E_EDGES) {
                float a = sAtt[el * 4 + slot];
                float l = a > 0.f ? a : 0.2f * a;       // leaky_relu
                float e = __expf(-l);
                ebuf[(size_t)edge * 4 + slot] = e;
                int node = (slot < NHEADS) ? coli[edge] : rowi[edge];
                unsafeAtomicAdd(&denom[(size_t)node * 4 + slot], e);
            }
        }
    }
}

// -------------------- dense WMMA GEMM: out = A @ W^T (+bias) (+add) --------
// Safe to run in-place (out == A): each block only touches its own 32 rows.
__global__ __launch_bounds__(64) void dense_gemm(
    const float* __restrict__ A, int rows, int K, int Kpad, int KC,
    const unsigned short* __restrict__ wfrag, int NT, int Dout,
    const float* __restrict__ bias, const float* __restrict__ addsrc,
    float* __restrict__ out) {
    __shared__ __align__(32) unsigned short sA[2 * 7 * 512];    // up to K=224
    __shared__ __align__(32) unsigned short sB[7 * 512];
    int tid = threadIdx.x, lane = tid & 31, wv = tid >> 5;
    int rbase = blockIdx.x * 32;

    for (int i = tid; i < 32 * Kpad; i += 64) {
        int el = i / Kpad, k = i - el * Kpad;
        int row = rbase + el;
        float v = (row < rows && k < K) ? A[(size_t)row * K + k] : 0.f;
        int tile = el >> 4, m = el & 15, kc = k >> 5, kk = k & 31;
        int hi = (kk >> 4) & 1;
        int ls = m + (hi << 4);
        int j  = kk & 15;
        sA[((tile * KC + kc) << 9) + (ls << 4) + j] = f2bf(v);
    }
    __syncthreads();

    int t = wv;
    for (int nt = 0; nt < NT; ++nt) {
        {
            const unsigned short* gsrc = &wfrag[((size_t)nt * KC) << 9];
            for (int c = tid; c < KC * 64; c += 64) {
                unsigned lds = (unsigned)(unsigned long long)&sB[c * 8];
                async_copy_b128(lds, gsrc + c * 8);
            }
            wait_asynccnt0();
        }
        __syncthreads();

        v8f acc = {0.f, 0.f, 0.f, 0.f, 0.f, 0.f, 0.f, 0.f};
        for (int kc = 0; kc < KC; ++kc) {
            v16bf a = *(const v16bf*)&sA[((t * KC + kc) << 9) + (lane << 4)];
            v16bf b = *(const v16bf*)&sB[(kc << 9) + (lane << 4)];
            acc = __builtin_amdgcn_wmma_f32_16x16x32_bf16(
                false, a, false, b, (short)0, acc, false, false);
        }
        int n  = nt * 16 + (lane & 15);
        int mh = (lane >> 4) << 3;
        if (n < Dout) {
            for (int r = 0; r < 8; ++r) {
                int row = rbase + t * 16 + r + mh;
                if (row < rows) {
                    float v = acc[r];
                    if (bias)   v += bias[n];
                    if (addsrc) v += addsrc[(size_t)row * Dout + n];
                    out[(size_t)row * Dout + n] = v;
                }
            }
        }
        __syncthreads();
    }
}

// -------------------- per-node GAT finalize: elu + per-head l2norm ---------
__global__ __launch_bounds__(256) void gat_finalize(float* __restrict__ h, int OUT) {
    __shared__ float s[2];
    int node = blockIdx.x, t = threadIdx.x;
    if (t < 2) s[t] = 0.f;
    __syncthreads();
    float e = 0.f; int head = 0;
    if (t < D_) {
        float v = h[(size_t)node * D_ + t];
        e = v > 0.f ? v : (__expf(v) - 1.f);
        head = t / OUT;
        atomicAdd(&s[head], e * e);
    }
    __syncthreads();
    if (t < D_)
        h[(size_t)node * D_ + t] = e / fmaxf(sqrtf(s[head]), 1e-12f);
}

// -------------------- merge layer: lambda-gate + elu + l2norm --------------
__global__ __launch_bounds__(256) void merge_layer_k(
    const float* __restrict__ hi, const float* __restrict__ ho,
    const float* __restrict__ Wl, const float* __restrict__ bl,
    float* __restrict__ out) {
    __shared__ float sd[2];     // [0]=gate dot, [1]=sumsq
    int node = blockIdx.x, t = threadIdx.x;
    if (t < 2) sd[t] = 0.f;
    __syncthreads();
    float p = 0.f;
    for (int k = t; k < 2 * D_; k += 256) {
        float v = (k < D_) ? hi[(size_t)node * D_ + k]
                           : ho[(size_t)node * D_ + (k - D_)];
        p += v * Wl[k];
    }
    atomicAdd(&sd[0], p);
    __syncthreads();
    float lam = 1.f / (1.f + __expf(-(sd[0] + bl[0])));
    float e = 0.f;
    if (t < D_) {
        float v = lam * hi[(size_t)node * D_ + t] +
                  (1.f - lam) * ho[(size_t)node * D_ + t];
        e = v > 0.f ? v : (__expf(v) - 1.f);
        atomicAdd(&sd[1], e * e);
    }
    __syncthreads();
    if (t < D_)
        out[(size_t)node * D_ + t] = e / fmaxf(sqrtf(sd[1]), 1e-12f);
}

// -------------------- relation layer: segment-sum of h_ijk over edge_type --
__global__ void gedge_accum(const int* __restrict__ rowi, const int* __restrict__ coli,
                            const int* __restrict__ et,
                            const float* __restrict__ xn, const float* __restrict__ g,
                            float* __restrict__ gacc) {
    long total = (long)E_EDGES * 300;
    for (long i = (long)blockIdx.x * blockDim.x + threadIdx.x; i < total;
         i += (long)gridDim.x * blockDim.x) {
        int e = (int)(i / 300), k = (int)(i % 300);
        float v;
        if (k < 100)       v = xn[(size_t)rowi[e] * 100 + k];
        else if (k < 200)  v = xn[(size_t)coli[e] * 100 + (k - 100)];
        else               v = g[(size_t)et[e] * 100 + (k - 200)];
        unsafeAtomicAdd(&gacc[(size_t)et[e] * 300 + k], v);
    }
}

__global__ __launch_bounds__(256) void relation_layer(
    const float* __restrict__ g, const float* __restrict__ gacc,
    const float* __restrict__ Wr, const float* __restrict__ br,
    const float* __restrict__ Wf, const float* __restrict__ bf_,
    float* __restrict__ gp) {
    __shared__ float s;
    int m = blockIdx.x, t = threadIdx.x;
    if (t == 0) s = 0.f;
    __syncthreads();
    float p = 0.f;
    for (int k = t; k < 300; k += 256) { float v = gacc[(size_t)m * 300 + k]; p += v * v; }
    atomicAdd(&s, p);
    __syncthreads();
    float inv = 1.f / fmaxf(sqrtf(s), 1e-12f);
    if (t < D_) {
        int d = t;
        float acc = br[d] + bf_[d];
        for (int k = 0; k < 100; ++k)
            acc += g[(size_t)m * 100 + k] * Wr[(size_t)d * 100 + k];
        for (int k = 0; k < 300; ++k) {
            float v = gacc[(size_t)m * 300 + k] * inv;
            float e = v > 0.f ? v : (__expf(v) - 1.f);
            acc += e * Wf[(size_t)d * 300 + k];
        }
        gp[(size_t)m * D_ + d] = acc;
    }
}

// ---------------------------------------------------------------------------
extern "C" void kernel_launch(void* const* d_in, const int* in_sizes, int n_in,
                              void* d_out, int out_size, void* d_ws, size_t ws_size,
                              hipStream_t stream) {
    const int* edge_idx = (const int*)d_in[0];
    const int* rowi = edge_idx;
    const int* coli = edge_idx + E_EDGES;
    const int* etype = (const int*)d_in[1];
    const float* x  = (const float*)d_in[5];
    const float* g  = (const float*)d_in[6];
    const float* W_in1  = (const float*)d_in[7];
    const float* att_in1 = (const float*)d_in[8];
    const float* W_out1 = (const float*)d_in[9];
    const float* att_out1 = (const float*)d_in[10];
    const float* Wi1 = (const float*)d_in[11]; const float* bi1 = (const float*)d_in[12];
    const float* Wo1 = (const float*)d_in[13]; const float* bo1 = (const float*)d_in[14];
    const float* Wl1 = (const float*)d_in[15]; const float* bl1 = (const float*)d_in[16];
    const float* Wr  = (const float*)d_in[17]; const float* br  = (const float*)d_in[18];
    const float* Wf  = (const float*)d_in[19]; const float* bf_ = (const float*)d_in[20];
    const float* W_in2  = (const float*)d_in[21];
    const float* att_in2 = (const float*)d_in[22];
    const float* W_out2 = (const float*)d_in[23];
    const float* att_out2 = (const float*)d_in[24];
    const float* Wi2 = (const float*)d_in[25]; const float* bi2 = (const float*)d_in[26];
    const float* Wo2 = (const float*)d_in[27]; const float* bo2 = (const float*)d_in[28];
    const float* Wl2 = (const float*)d_in[29]; const float* bl2 = (const float*)d_in[30];
    const float* W_ent = (const float*)d_in[31];

    // ---- workspace carving ----
    char* base = (char*)d_ws;
    size_t off = 0;
    auto carve = [&](size_t bytes) -> char* {
        char* p = base + off;
        off = (off + bytes + 255) & ~(size_t)255;
        return p;
    };
    float* xn    = (float*)carve((size_t)N_NODES * XS_ * 4);
    float* denom = (float*)carve((size_t)N_NODES * 4 * 4);
    float* ebuf  = (float*)carve((size_t)E_EDGES * 4 * 4);
    float* hin   = (float*)carve((size_t)N_NODES * D_ * 4);
    float* hout  = (float*)carve((size_t)N_NODES * D_ * 4);
    float* h1    = (float*)carve((size_t)N_NODES * D_ * 4);
    float* gacc  = (float*)carve((size_t)M_REL * 300 * 4);
    float* gp    = (float*)carve((size_t)M_REL * D_ * 4);
    unsigned short* fW1   = (unsigned short*)carve((size_t)25 * 10 * 512 * 2);
    unsigned short* fW2   = (unsigned short*)carve((size_t)25 * 20 * 512 * 2);
    unsigned short* fWi1  = (unsigned short*)carve((size_t)13 * 7 * 512 * 2);
    unsigned short* fWo1  = (unsigned short*)carve((size_t)13 * 7 * 512 * 2);
    unsigned short* fWi2  = (unsigned short*)carve((size_t)13 * 7 * 512 * 2);
    unsigned short* fWo2  = (unsigned short*)carve((size_t)13 * 7 * 512 * 2);
    unsigned short* fWent = (unsigned short*)carve((size_t)13 * 4 * 512 * 2);

    const int EB = (E_EDGES + 31) / 32;        // edge blocks (64 threads each)
    const int RB = (N_NODES + 31) / 32;        // row blocks for dense GEMMs

    // ---- phase 0: normalize x, pack weights ----
    l2norm_rows<<<N_NODES, 128, 0, stream>>>(x, xn, XS_);
    pack_bfrag<<<512, 256, 0, stream>>>(W_in1, W_out1, 200, 400, 300, 25, 10, fW1);
    pack_bfrag<<<512, 256, 0, stream>>>(W_in2, W_out2, 200, 400, 600, 25, 20, fW2);
    pack_bfrag<<<256, 256, 0, stream>>>(Wi1, Wi1, 200, 200, 200, 13, 7, fWi1);
    pack_bfrag<<<256, 256, 0, stream>>>(Wo1, Wo1, 200, 200, 200, 13, 7, fWo1);
    pack_bfrag<<<256, 256, 0, stream>>>(Wi2, Wi2, 200, 200, 200, 13, 7, fWi2);
    pack_bfrag<<<256, 256, 0, stream>>>(Wo2, Wo2, 200, 200, 200, 13, 7, fWo2);
    pack_bfrag<<<256, 256, 0, stream>>>(W_ent, W_ent, 200, 200, 100, 13, 4, fWent);

    // ---- hop 1 ----
    hipMemsetAsync(denom, 0, (size_t)N_NODES * 4 * 4, stream);
    hipMemsetAsync(hin,   0, (size_t)N_NODES * D_ * 4, stream);
    hipMemsetAsync(hout,  0, (size_t)N_NODES * D_ * 4, stream);
    hipMemsetAsync(gacc,  0, (size_t)M_REL * 300 * 4, stream);

    edge_gat_pass<<<EB, 64, 0, stream>>>(0, rowi, coli, etype,
        xn, 100, xn, 100, g, 100, 320, 10, fW1, 25, 100, 2,
        att_in1, att_out1, ebuf, denom, hin, hout);
    edge_gat_pass<<<EB, 64, 0, stream>>>(1, rowi, coli, etype,
        xn, 100, xn, 100, g, 100, 320, 10, fW1, 25, 100, 2,
        att_in1, att_out1, ebuf, denom, hin, hout);

    gedge_accum<<<2048, 256, 0, stream>>>(rowi, coli, etype, xn, g, gacc);
    relation_layer<<<M_REL, 256, 0, stream>>>(g, gacc, Wr, br, Wf, bf_, gp);

    gat_finalize<<<N_NODES, 256, 0, stream>>>(hin, 100);
    gat_finalize<<<N_NODES, 256, 0, stream>>>(hout, 100);
    dense_gemm<<<RB, 64, 0, stream>>>(hin, N_NODES, 200, 224, 7, fWi1, 13, 200,
                                      bi1, nullptr, hin);   // in-place hi
    dense_gemm<<<RB, 64, 0, stream>>>(hout, N_NODES, 200, 224, 7, fWo1, 13, 200,
                                      bo1, nullptr, hout);  // in-place ho
    merge_layer_k<<<N_NODES, 256, 0, stream>>>(hin, hout, Wl1, bl1, h1);

    // ---- hop 2 ----
    hipMemsetAsync(denom, 0, (size_t)N_NODES * 4 * 4, stream);
    hipMemsetAsync(hin,   0, (size_t)N_NODES * D_ * 4, stream);
    hipMemsetAsync(hout,  0, (size_t)N_NODES * D_ * 4, stream);

    edge_gat_pass<<<EB, 64, 0, stream>>>(0, rowi, coli, etype,
        h1, 200, h1, 200, gp, 200, 640, 20, fW2, 25, 200, 1,
        att_in2, att_out2, ebuf, denom, hin, hout);
    edge_gat_pass<<<EB, 64, 0, stream>>>(1, rowi, coli, etype,
        h1, 200, h1, 200, gp, 200, 640, 20, fW2, 25, 200, 1,
        att_in2, att_out2, ebuf, denom, hin, hout);

    gat_finalize<<<N_NODES, 256, 0, stream>>>(hin, 200);
    gat_finalize<<<N_NODES, 256, 0, stream>>>(hout, 200);
    dense_gemm<<<RB, 64, 0, stream>>>(hin, N_NODES, 200, 224, 7, fWi2, 13, 200,
                                      bi2, nullptr, hin);
    dense_gemm<<<RB, 64, 0, stream>>>(hout, N_NODES, 200, 224, 7, fWo2, 13, 200,
                                      bo2, nullptr, hout);
    merge_layer_k<<<N_NODES, 256, 0, stream>>>(hin, hout, Wl2, bl2, h1); // h1 <- h2

    // ---- final: d_out = xn @ W_ent^T + h2 ----
    dense_gemm<<<RB, 64, 0, stream>>>(xn, N_NODES, 100, 128, 4, fWent, 13, 200,
                                      nullptr, h1, (float*)d_out);

    (void)in_sizes; (void)n_in; (void)out_size; (void)ws_size;
}
